// TrendAwareAttentionLayer_73323681677864
// MI455X (gfx1250) — matrix operand order repaired and checked by reading, not compiled, not run on hardware
//
#include <hip/hip_runtime.h>
#include <hip/hip_bf16.h>

// ---------- problem constants ----------
#define BB   8
#define TT   1024
#define DD   1024
#define HH   16
#define DKK  64
#define KTAP 3

typedef __attribute__((ext_vector_type(16))) __bf16 v16bf;
typedef __attribute__((ext_vector_type(8)))  float  v8f;

// ---------------------------------------------------------------------------
// Fragment loaders (wave32); A and B share one contiguous k order, k = 16*(lane>>4) + i.
// A (16x32 bf16) from row-major [M][K]: lane L<16 row L, halves K=kb..kb+15, kb=0;
//                                       lane L>=16 same row, kb=16.  (one 32B run)
// B (32x16 bf16) given K-contiguous per column ([N][K] storage): lane holds column lane&15,
//                                       halves K=(lane>>4)*16 + i. (one 32B run)
// C/D (16x16 f32): vgpr r, lane: M = r + 8*(lane>=16), N = lane&15.
// ---------------------------------------------------------------------------
__device__ __forceinline__ v16bf load_a_frag(const __bf16* __restrict__ base,
                                             int stride, int rowBase, int lane) {
  int row = rowBase + (lane & 15);
  int kb  = (lane >> 4) * 16;
  const __bf16* p = base + (size_t)row * stride + kb;
  v16bf a;
#pragma unroll
  for (int i = 0; i < 8; ++i) a[i] = p[i];
#pragma unroll
  for (int i = 0; i < 8; ++i) a[8 + i] = p[8 + i];
  return a;
}

__device__ __forceinline__ v16bf load_bt_frag(const __bf16* __restrict__ base,
                                              int stride, int colBase, int lane) {
  int col = colBase + (lane & 15);
  int kh  = (lane >> 4) * 16;
  const __bf16* p = base + (size_t)col * stride + kh;
  v16bf b;
#pragma unroll
  for (int i = 0; i < 16; ++i) b[i] = p[i];
  return b;
}

__device__ __forceinline__ v8f wmma_bf16(v16bf a, v16bf b, v8f c) {
  return __builtin_amdgcn_wmma_f32_16x16x32_bf16(false, a, false, b,
                                                 (short)0, c, false, false);
}

// ---------------------------------------------------------------------------
// elementwise f32 -> bf16
// ---------------------------------------------------------------------------
__global__ void cvt_bf16_kernel(const float* __restrict__ s,
                                __bf16* __restrict__ d, long long n) {
  long long i = (long long)blockIdx.x * blockDim.x + threadIdx.x;
  long long stride = (long long)gridDim.x * blockDim.x;
  for (; i < n; i += stride) d[i] = (__bf16)s[i];
}

// w[o][i][k] (row-major (Dd,Dd,Kt)) -> wtT[k][o][i] bf16 (B^T layout: K-contiguous per out col).
// Kt==1 degenerates to plain f32->bf16 copy of W[o][i].
__global__ void wt_transpose_kernel(const float* __restrict__ w,
                                    __bf16* __restrict__ wt, int Dd, int Kt) {
  long long idx = (long long)blockIdx.x * blockDim.x + threadIdx.x;
  long long total = (long long)Dd * Dd * Kt;
  if (idx >= total) return;
  int i = (int)(idx % Dd);
  long long rem = idx / Dd;
  int o = (int)(rem % Dd);
  int k = (int)(rem / Dd);
  wt[idx] = (__bf16)w[((size_t)o * Dd + i) * Kt + k];
}

// ---------------------------------------------------------------------------
// Generic conv/GEMM: out[m][n] = bias[n] + sum_tap sum_i A[row(m,tap)][i] * WtT[tap][n][i]
// row(m,tap): m = b*T + t, source row t + tap - ntaps/2, zero outside [0,T).
// Block tile 128x128; 8 waves, each a 32x64 macro-tile (2x4 WMMA frags).
// Register-staged pipeline: next tile's global loads overlap current tile's WMMAs.
// Output: f32 [M][N] and/or bf16 [M][N] and/or bf16 transposed [B][N][T].
// ---------------------------------------------------------------------------
__global__ __launch_bounds__(256) void conv_gemm_bf16(
    const __bf16* __restrict__ A,    // [M][Kd]
    const __bf16* __restrict__ WtT,  // [ntaps][N][Kd]
    const float*  __restrict__ bias,
    __bf16* __restrict__ outBf,      // [M][N] or nullptr
    __bf16* __restrict__ outBfT,     // [B][N][T] or nullptr
    float*  __restrict__ outF,       // [M][N] or nullptr
    int M, int N, int Kd, int ntaps, int Tdim) {
  const int tid  = threadIdx.x;
  const int lane = tid & 31;
  const int wid  = tid >> 5;
  const int wm   = wid >> 1;  // 0..3
  const int wn   = wid & 1;   // 0..1
  const int rowBase = blockIdx.x * 128;
  const int colBase = blockIdx.y * 128;

  __shared__ __attribute__((aligned(16))) __bf16 sA[128 * 32];   // [m][k]
  __shared__ __attribute__((aligned(16))) __bf16 sBt[128 * 32];  // [n][k]

  const v8f vzero = {0.f, 0.f, 0.f, 0.f, 0.f, 0.f, 0.f, 0.f};
  v8f acc[2][4];
#pragma unroll
  for (int i = 0; i < 2; ++i)
#pragma unroll
    for (int j = 0; j < 4; ++j) acc[i][j] = vzero;

  const int center = ntaps >> 1;
  const int sr = tid >> 1, scb = (tid & 1) * 16;  // staging: 16 halves/thread (both tiles)

  const int g  = rowBase + sr;
  const int t0 = g % Tdim;
  const int b0 = g / Tdim;

  const int kPerTap = Kd >> 5;
  const int ksteps  = ntaps * kPerTap;

  int ldTap = 0, ldKk = 0;  // next step to load
  uint4 ra0, ra1, rb0, rb1;
  {  // prologue: load step 0
    int ts = t0 + ldTap - center;
    bool valid = (ts >= 0) && (ts < Tdim);
    ra0 = make_uint4(0, 0, 0, 0);
    ra1 = ra0;
    if (valid) {
      const uint4* srcA = (const uint4*)(A + ((size_t)b0 * Tdim + ts) * Kd + ldKk + scb);
      ra0 = srcA[0];
      ra1 = srcA[1];
      __builtin_prefetch(srcA + 2, 0, 1);
    }
    const uint4* srcB =
        (const uint4*)(WtT + ((size_t)ldTap * N + colBase + sr) * Kd + ldKk + scb);
    rb0 = srcB[0];
    rb1 = srcB[1];
    ldKk += 32;
    if (ldKk == Kd) { ldKk = 0; ++ldTap; }
  }

  for (int step = 0; step < ksteps; ++step) {
    {  // commit staged registers to LDS
      uint4* dA = (uint4*)(sA + sr * 32 + scb);
      dA[0] = ra0;
      dA[1] = ra1;
      uint4* dB = (uint4*)(sBt + sr * 32 + scb);
      dB[0] = rb0;
      dB[1] = rb1;
    }
    __syncthreads();
    if (step + 1 < ksteps) {  // issue next tile's global loads (overlap with WMMAs)
      int ts = t0 + ldTap - center;
      bool valid = (ts >= 0) && (ts < Tdim);
      ra0 = make_uint4(0, 0, 0, 0);
      ra1 = ra0;
      if (valid) {
        const uint4* srcA =
            (const uint4*)(A + ((size_t)b0 * Tdim + ts) * Kd + ldKk + scb);
        ra0 = srcA[0];
        ra1 = srcA[1];
        __builtin_prefetch(srcA + 2, 0, 1);
      }
      const uint4* srcB =
          (const uint4*)(WtT + ((size_t)ldTap * N + colBase + sr) * Kd + ldKk + scb);
      rb0 = srcB[0];
      rb1 = srcB[1];
      ldKk += 32;
      if (ldKk == Kd) { ldKk = 0; ++ldTap; }
    }
#pragma unroll
    for (int i = 0; i < 2; ++i) {
      v16bf af = load_a_frag(sA, 32, wm * 32 + i * 16, lane);
#pragma unroll
      for (int j = 0; j < 4; ++j) {
        v16bf bf = load_bt_frag(sBt, 32, wn * 64 + j * 16, lane);
        acc[i][j] = wmma_bf16(af, bf, acc[i][j]);
      }
    }
    __syncthreads();
  }

  // epilogue
  const int n0 = lane & 15;
  const int mo = (lane >> 4) * 8;
#pragma unroll
  for (int i = 0; i < 2; ++i) {
    int rTile = rowBase + wm * 32 + i * 16;
#pragma unroll
    for (int j = 0; j < 4; ++j) {
      int col = colBase + wn * 64 + j * 16 + n0;
      float bv = bias ? bias[col] : 0.0f;
#pragma unroll
      for (int r = 0; r < 8; ++r) {
        int row = rTile + mo + r;
        float v = acc[i][j][r] + bv;
        if (outF)  outF[(size_t)row * N + col] = v;
        if (outBf) outBf[(size_t)row * N + col] = (__bf16)v;
        if (outBfT) {
          int bb = row / Tdim, t = row % Tdim;
          outBfT[((size_t)bb * N + col) * Tdim + t] = (__bf16)v;
        }
      }
    }
  }
}

// ---------------------------------------------------------------------------
// scores + online softmax stats.
// One block owns a 128-row q-strip of one (b,h); loops all 16 column tiles.
// Writes raw masked scores (f32, -inf for masked) and per-row (max, sum_exp).
// ---------------------------------------------------------------------------
__global__ __launch_bounds__(256) void scores_kernel(
    const __bf16* __restrict__ Qb, const __bf16* __restrict__ Kb,
    const unsigned char* __restrict__ kpad,  // [B][T] bool
    float* __restrict__ S,                   // [BH][T][T] raw masked scores
    float2* __restrict__ stats) {            // [BH][T] (rowMax, rowSumExp)
  const int bh = blockIdx.z;
  const int b = bh / HH, h = bh % HH;
  const int qBase = blockIdx.x * 128;
  const int tid = threadIdx.x, lane = tid & 31, wid = tid >> 5;
  const int wm = wid >> 1, wn = wid & 1;
  const int n0 = lane & 15;
  const int mo = (lane >> 4) * 8;
  const float NEGINF = -__builtin_inff();

  // hoist Q fragments: [ks2][i]
  v16bf af[2][2];
#pragma unroll
  for (int ks2 = 0; ks2 < 2; ++ks2)
#pragma unroll
    for (int i = 0; i < 2; ++i)
      af[ks2][i] = load_a_frag(
          Qb + ((size_t)b * TT + qBase + wm * 32 + i * 16) * DD + h * DKK + ks2 * 32,
          DD, 0, lane);

  float runM[2][8], runS[2][8];
#pragma unroll
  for (int i = 0; i < 2; ++i)
#pragma unroll
    for (int r = 0; r < 8; ++r) { runM[i][r] = NEGINF; runS[i][r] = 0.f; }

  const v8f vzero = {0.f, 0.f, 0.f, 0.f, 0.f, 0.f, 0.f, 0.f};

  for (int sb = 0; sb < TT; sb += 64) {
    v8f acc[2][2];
#pragma unroll
    for (int i = 0; i < 2; ++i)
#pragma unroll
      for (int j = 0; j < 2; ++j) acc[i][j] = vzero;

#pragma unroll
    for (int j = 0; j < 2; ++j) {
#pragma unroll
      for (int ks2 = 0; ks2 < 2; ++ks2) {
        v16bf bf = load_bt_frag(
            Kb + ((size_t)b * TT + sb + wn * 32 + j * 16) * DD + h * DKK + ks2 * 32,
            DD, 0, lane);
#pragma unroll
        for (int i = 0; i < 2; ++i) acc[i][j] = wmma_bf16(af[ks2][i], bf, acc[i][j]);
      }
    }

    // epilogue: mask, write raw scores, online stats
#pragma unroll
    for (int i = 0; i < 2; ++i) {
      int qT = qBase + wm * 32 + i * 16;
#pragma unroll
      for (int j = 0; j < 2; ++j) {
        int s = sb + wn * 32 + j * 16 + n0;
        bool pad = kpad[(size_t)b * TT + s] != 0;
        float x[8];
#pragma unroll
        for (int r = 0; r < 8; ++r) {
          int q = qT + mo + r;
          bool mk = pad || (s > q);  // causal: strict upper triangle masked
          x[r] = mk ? NEGINF : acc[i][j][r] * 0.125f;
          S[((size_t)bh * TT + q) * TT + s] = x[r];
        }
#pragma unroll
        for (int r = 0; r < 8; ++r) {
          float tm = x[r];
#pragma unroll
          for (int m = 1; m < 16; m <<= 1) tm = fmaxf(tm, __shfl_xor(tm, m, 32));
          float nm = fmaxf(runM[i][r], tm);
          float e = (x[r] == NEGINF) ? 0.f : __expf(x[r] - nm);
#pragma unroll
          for (int m = 1; m < 16; m <<= 1) e += __shfl_xor(e, m, 32);
          float sc = (runM[i][r] == nm) ? 1.f : __expf(runM[i][r] - nm);
          runS[i][r] = runS[i][r] * sc + e;
          runM[i][r] = nm;
        }
      }
    }
  }

  // combine the two column-half waves per row
  __shared__ float shM[2][128], shS[2][128];
  if (n0 == 0) {
#pragma unroll
    for (int i = 0; i < 2; ++i)
#pragma unroll
      for (int r = 0; r < 8; ++r) {
        int rloc = wm * 32 + i * 16 + mo + r;
        shM[wn][rloc] = runM[i][r];
        shS[wn][rloc] = runS[i][r];
      }
  }
  __syncthreads();
  if (tid < 128) {
    float m0 = shM[0][tid], m1 = shM[1][tid];
    float s0 = shS[0][tid], s1 = shS[1][tid];
    float m = fmaxf(m0, m1);
    float g0 = (m0 == m) ? 1.f : __expf(m0 - m);
    float g1 = (m1 == m) ? 1.f : __expf(m1 - m);
    stats[(size_t)bh * TT + qBase + tid] = make_float2(m, s0 * g0 + s1 * g1);
  }
}

// ---------------------------------------------------------------------------
// PV: normalize weights on the fly (write f32 attn output in place) and
// compute out[b,q,h*64+d] = sum_s softmax(q,s) * V[b,s,h*64+d] via WMMA.
// Vt is the transposed value projection [B][D][T] so B-frags are contiguous.
// Register-staged pipeline like conv_gemm.
// ---------------------------------------------------------------------------
__global__ __launch_bounds__(256) void pv_kernel(float* __restrict__ S,
                                                 const float2* __restrict__ stats,
                                                 const __bf16* __restrict__ Vt,
                                                 __bf16* __restrict__ Ob) {
  const int bh = blockIdx.z;
  const int b = bh / HH, h = bh % HH;
  const int qBase = blockIdx.x * 128;
  const int tid = threadIdx.x, lane = tid & 31, wid = tid >> 5;
  const int wm = wid >> 1, wn = wid & 1;

  __shared__ __attribute__((aligned(16))) __bf16 sA[128 * 32];  // [q][s]
  __shared__ __attribute__((aligned(16))) __bf16 sBt[64 * 32];  // [dk][s]

  const v8f vzero = {0.f, 0.f, 0.f, 0.f, 0.f, 0.f, 0.f, 0.f};
  v8f acc[2][2];
#pragma unroll
  for (int i = 0; i < 2; ++i)
#pragma unroll
    for (int j = 0; j < 2; ++j) acc[i][j] = vzero;

  const int ar = tid >> 1, acb = (tid & 1) * 16;  // A staging: one row, 16 f32/thread
  const int bn = tid >> 2, bcb = (tid & 3) * 8;   // B staging: 8 halves/thread
  float* wRow = S + ((size_t)bh * TT + qBase + ar) * TT;
  const float2 st = stats[(size_t)bh * TT + qBase + ar];
  const float rowM = st.x;
  const float invS = 1.0f / st.y;
  const __bf16* vRow = Vt + ((size_t)b * DD + h * DKK + bn) * TT;

  float4 rf[4];
  uint4 rb;
  {  // prologue: load step 0
    const float4* srcA = (const float4*)(wRow + acb);
#pragma unroll
    for (int j = 0; j < 4; ++j) rf[j] = srcA[j];
    rb = *(const uint4*)(vRow + bcb);
  }

  for (int kk = 0; kk < TT; kk += 32) {
    {  // normalize + commit to LDS, and materialize normalized weights (output)
      float* wb = wRow + kk + acb;
      __bf16* dst = sA + ar * 32 + acb;
#pragma unroll
      for (int i = 0; i < 16; ++i) {
        float xv = ((const float*)rf)[i];
        float w = (xv < -3.0e38f) ? 0.f : __expf(xv - rowM) * invS;
        wb[i] = w;
        dst[i] = (__bf16)w;
      }
      *(uint4*)(sBt + bn * 32 + bcb) = rb;
    }
    __syncthreads();
    if (kk + 32 < TT) {  // next tile's global loads overlap WMMAs
      const float4* srcA = (const float4*)(wRow + kk + 32 + acb);
#pragma unroll
      for (int j = 0; j < 4; ++j) rf[j] = srcA[j];
      rb = *(const uint4*)(vRow + kk + 32 + bcb);
      __builtin_prefetch(wRow + kk + 64 + acb, 0, 1);
    }
#pragma unroll
    for (int i = 0; i < 2; ++i) {
      v16bf af = load_a_frag(sA, 32, wm * 32 + i * 16, lane);
#pragma unroll
      for (int j = 0; j < 2; ++j) {
        v16bf bf = load_bt_frag(sBt, 32, wn * 32 + j * 16, lane);
        acc[i][j] = wmma_bf16(af, bf, acc[i][j]);
      }
    }
    __syncthreads();
  }

  const int n0 = lane & 15;
  const int mo = (lane >> 4) * 8;
#pragma unroll
  for (int i = 0; i < 2; ++i) {
    int qT = qBase + wm * 32 + i * 16;
#pragma unroll
    for (int j = 0; j < 2; ++j) {
      int col = wn * 32 + j * 16 + n0;  // 0..63 within head
#pragma unroll
      for (int r = 0; r < 8; ++r) {
        int q = qT + mo + r;
        Ob[((size_t)b * TT + q) * DD + h * DKK + col] = (__bf16)acc[i][j][r];
      }
    }
  }
}

// ---------------------------------------------------------------------------
extern "C" void kernel_launch(void* const* d_in, const int* in_sizes, int n_in,
                              void* d_out, int out_size, void* d_ws,
                              size_t ws_size, hipStream_t stream) {
  const float* q        = (const float*)d_in[0];
  const float* k        = (const float*)d_in[1];
  const float* v        = (const float*)d_in[2];
  // d_in[3] = attn_mask (causal triu, applied analytically)
  const unsigned char* kpad = (const unsigned char*)d_in[4];  // [B][T] bool
  const float* conv_q_w = (const float*)d_in[5];
  const float* conv_q_b = (const float*)d_in[6];
  const float* conv_k_w = (const float*)d_in[7];
  const float* conv_k_b = (const float*)d_in[8];
  const float* v_w      = (const float*)d_in[9];
  const float* v_b      = (const float*)d_in[10];
  const float* out_w    = (const float*)d_in[11];
  const float* out_b    = (const float*)d_in[12];

  float* out   = (float*)d_out;               // [B,T,D]
  float* attnW = out + (size_t)BB * TT * DD;  // [B,H,T,T]

  const size_t MiB = 1ull << 20;
  char* ws = (char*)d_ws;
  __bf16* q_bf    = (__bf16*)(ws + 0 * MiB);    // 16 MiB
  __bf16* k_bf    = (__bf16*)(ws + 16 * MiB);   // 16 MiB
  __bf16* v_bf    = (__bf16*)(ws + 32 * MiB);   // 16 MiB
  __bf16* wq_t    = (__bf16*)(ws + 48 * MiB);   // 6 MiB  [3][1024][1024] (B^T)
  __bf16* wk_t    = (__bf16*)(ws + 54 * MiB);   // 6 MiB
  __bf16* vw_t    = (__bf16*)(ws + 60 * MiB);   // 2 MiB  [1024][1024]
  __bf16* ow_t    = (__bf16*)(ws + 62 * MiB);   // 2 MiB
  __bf16* qout_bf = (__bf16*)(ws + 64 * MiB);   // 16 MiB [B,T,D]
  __bf16* kout_bf = (__bf16*)(ws + 80 * MiB);   // 16 MiB [B,T,D]
  __bf16* vout_tr = (__bf16*)(ws + 96 * MiB);   // 16 MiB [B,D,T] (transposed!)
  __bf16* aout_bf = (__bf16*)(ws + 112 * MiB);  // 16 MiB [B,T,D]
  float2* stats   = (float2*)(ws + 128 * MiB);  // 1 MiB  [BH][T]

  const long long nElem = (long long)BB * TT * DD;  // 8388608

  // 1) precision conversion + weight transposes
  cvt_bf16_kernel<<<4096, 256, 0, stream>>>(q, q_bf, nElem);
  cvt_bf16_kernel<<<4096, 256, 0, stream>>>(k, k_bf, nElem);
  cvt_bf16_kernel<<<4096, 256, 0, stream>>>(v, v_bf, nElem);
  wt_transpose_kernel<<<(DD * DD * KTAP) / 256, 256, 0, stream>>>(conv_q_w, wq_t, DD, KTAP);
  wt_transpose_kernel<<<(DD * DD * KTAP) / 256, 256, 0, stream>>>(conv_k_w, wk_t, DD, KTAP);
  wt_transpose_kernel<<<(DD * DD) / 256, 256, 0, stream>>>(v_w, vw_t, DD, 1);
  wt_transpose_kernel<<<(DD * DD) / 256, 256, 0, stream>>>(out_w, ow_t, DD, 1);

  // 2) projections (M=8192, N=1024, Kd=1024), block tile 128x128
  dim3 gGemm(8192 / 128, 1024 / 128);
  conv_gemm_bf16<<<gGemm, 256, 0, stream>>>(q_bf, wq_t, conv_q_b, qout_bf, nullptr,
                                            nullptr, 8192, 1024, 1024, KTAP, TT);
  conv_gemm_bf16<<<gGemm, 256, 0, stream>>>(k_bf, wk_t, conv_k_b, kout_bf, nullptr,
                                            nullptr, 8192, 1024, 1024, KTAP, TT);
  conv_gemm_bf16<<<gGemm, 256, 0, stream>>>(v_bf, vw_t, v_b, nullptr, vout_tr,
                                            nullptr, 8192, 1024, 1024, 1, TT);

  // 3) masked raw scores + online softmax stats -> d_out attn region + stats
  scores_kernel<<<dim3(TT / 128, 1, BB * HH), 256, 0, stream>>>(qout_bf, kout_bf,
                                                                kpad, attnW, stats);

  // 4) normalize-in-place + attn @ V, then output projection
  pv_kernel<<<dim3(TT / 128, 1, BB * HH), 256, 0, stream>>>(attnW, stats, vout_tr,
                                                            aout_bf);
  conv_gemm_bf16<<<gGemm, 256, 0, stream>>>(aout_bf, ow_t, out_b, nullptr, nullptr,
                                            out, 8192, 1024, 1024, 1, TT);
}
